// NeuralClickModel_46634754900495
// MI455X (gfx1250) — matrix operand order repaired and checked by reading, not compiled, not run on hardware
//
#include <hip/hip_runtime.h>

typedef __attribute__((ext_vector_type(16))) __bf16 v16bf;
typedef __attribute__((ext_vector_type(8)))  __bf16 v8bf;
typedef __attribute__((ext_vector_type(8)))  float  v8f;

#define DIM     256
#define SLATE_N 20
#define NROWS   4096      // B*T
#define NG      81920     // B*T*SLATE
#define THREE_D 768

// ---- WMMA fragment loaders (wave32, 16x16x32 bf16) -------------------------
// A matrix 16x32 (MxK), row-major source with leading dim ld:
// lane L: M = L&15 ; 16 contiguous K at offset (L>>4)*16 across VGPR0-7
__device__ __forceinline__ v16bf load_a_rm(const __bf16* __restrict__ base, int ld) {
  const int lane = threadIdx.x & 31;
  const int r  = lane & 15;
  const int kh = (lane >> 4) << 4;          // 0 or 16
  const __bf16* p0 = base + r * ld + kh;
  const __bf16* p1 = p0 + 8;
  v8bf lo = *(const v8bf*)p0;
  v8bf hi = *(const v8bf*)p1;
  v16bf f;
#pragma unroll
  for (int i = 0; i < 8; ++i) { f[i] = lo[i]; f[i + 8] = hi[i]; }
  return f;
}

// B matrix 32x16 (KxN). Source stored as Bᵀ row-major: rows = output columns.
// lane L: N = L&15 ; 16 contiguous K at offset (L>>4)*16 across VGPR0-7.
__device__ __forceinline__ v16bf load_b_rm(const __bf16* __restrict__ base, int ld) {
  const int lane = threadIdx.x & 31;
  const int n  = lane & 15;
  const int kb = (lane >> 4) << 4;          // 0 or 16
  const __bf16* p = base + n * ld + kb;
  v8bf lo = *(const v8bf*)p;
  v8bf hi = *(const v8bf*)(p + 8);
  v16bf f;
#pragma unroll
  for (int i = 0; i < 8; ++i) { f[i] = lo[i]; f[i + 8] = hi[i]; }
  return f;
}

__device__ __forceinline__ float sigmoidf_(float x) {
  return 1.0f / (1.0f + __expf(-x));
}

// ---- Kernel 1: convert weights to bf16 ------------------------------------
// W_ih [768][512] -> W1b[768][256] (cols 0..255), W2b[768][256] (cols 256..511)
// W_hh [768][256] -> Whhb
__global__ void k_prep_weights(const float* __restrict__ W_ih,
                               const float* __restrict__ W_hh,
                               __bf16* __restrict__ W1b,
                               __bf16* __restrict__ W2b,
                               __bf16* __restrict__ Whhb) {
  int i = blockIdx.x * blockDim.x + threadIdx.x;
  if (i >= THREE_D * DIM) return;
  int r = i / DIM, c = i % DIM;
  W1b[i]  = (__bf16)W_ih[r * (2 * DIM) + c];
  W2b[i]  = (__bf16)W_ih[r * (2 * DIM) + DIM + c];
  Whhb[i] = (__bf16)W_hh[i];
}

// ---- Kernel 2: big parallel GEMM ------------------------------------------
// G1[g][j] = (E @ W1ᵀ)[g][j] + b_ih[j] ; G2 = E @ W2ᵀ.  g = n*SLATE + s.
// One block = 16 rows of g, 8 waves, each wave does 6 N-tiles x 2 matrices.
__global__ void __launch_bounds__(256) k_gemm_gi(
    const int*   __restrict__ item_idxs,
    const float* __restrict__ item_table,
    const float* __restrict__ b_ih,
    const __bf16* __restrict__ W1b,
    const __bf16* __restrict__ W2b,
    __bf16* __restrict__ G1,
    __bf16* __restrict__ G2) {
  __shared__ __align__(16) __bf16 e_bf[16 * DIM];
  const int g0 = blockIdx.x * 16;

  // gather + f32->bf16: thread t loads 16 contiguous floats of one row
  {
    int r  = threadIdx.x >> 4;                // 0..15
    int cb = (threadIdx.x & 15) << 4;         // 0,16,...,240
    int idx = item_idxs[g0 + r];
    const float* src = item_table + (long)idx * DIM + cb;
#pragma unroll
    for (int c = 0; c < 16; ++c) e_bf[r * DIM + cb + c] = (__bf16)src[c];
  }
  __syncthreads();

  const int w    = threadIdx.x >> 5;
  const int lane = threadIdx.x & 31;
  const int half = lane >> 4;
  const int lc   = lane & 15;

  for (int t = 0; t < 6; ++t) {
    int nt = w + (t << 3);                    // N-tile 0..47
    v8f accA = {}; v8f accB = {};
#pragma unroll
    for (int k = 0; k < 8; ++k) {
      v16bf a  = load_a_rm(e_bf + k * 32, DIM);
      v16bf b1 = load_b_rm(W1b + nt * 16 * DIM + k * 32, DIM);
      v16bf b2 = load_b_rm(W2b + nt * 16 * DIM + k * 32, DIM);
      accA = __builtin_amdgcn_wmma_f32_16x16x32_bf16(false, a, false, b1,
                                                     (short)0, accA, false, false);
      accB = __builtin_amdgcn_wmma_f32_16x16x32_bf16(false, a, false, b2,
                                                     (short)0, accB, false, false);
    }
    int   col  = nt * 16 + lc;                // 0..767
    float bias = b_ih[col];
#pragma unroll
    for (int i = 0; i < 8; ++i) {
      int row = g0 + i + (half << 3);
      G1[(long)row * THREE_D + col] = (__bf16)(accA[i] + bias);
      G2[(long)row * THREE_D + col] = (__bf16)accB[i];
    }
  }
}

// ---- Kernel 3: sequential GRU recurrence ----------------------------------
// Block owns 16 rows n0..n0+15, iterates s=0..19 locally.
// Hidden state lives in C-layout fragments (2 unit-tiles of 16 per wave).
__global__ void __launch_bounds__(256) k_recurrence(
    const int*   __restrict__ user_idxs,
    const float* __restrict__ user_table,
    const float* __restrict__ b_hh,
    const float* __restrict__ W_out,
    const float* __restrict__ b_out,
    const __bf16* __restrict__ Whhb,
    const __bf16* __restrict__ G1,
    const __bf16* __restrict__ G2,
    float* __restrict__ out) {
  __shared__ __align__(16) __bf16 h_bf[16 * DIM];
  __shared__ float mbuf[16];
  __shared__ float ybuf[16];

  const int n0   = blockIdx.x * 16;
  const int w    = threadIdx.x >> 5;
  const int lane = threadIdx.x & 31;
  const int half = lane >> 4;
  const int lc   = lane & 15;
  const int ub0  = (w * 2) * 16;
  const int ub1  = (w * 2 + 1) * 16;

  // h0 = gathered user embedding, held in C-fragment layout
  v8f hfr0, hfr1;
#pragma unroll
  for (int i = 0; i < 8; ++i) {
    int row = i + (half << 3);
    long uidx = (long)user_idxs[n0 + row] * DIM;
    hfr0[i] = user_table[uidx + ub0 + lc];
    hfr1[i] = user_table[uidx + ub1 + lc];
  }
  if (threadIdx.x < 16) mbuf[threadIdx.x] = 1.0f;
  const float bout = b_out[0];

  for (int s = 0; s < SLATE_N; ++s) {
    // stage h (bf16) for the A operand; zero y accumulator
#pragma unroll
    for (int i = 0; i < 8; ++i) {
      int row = i + (half << 3);
      h_bf[row * DIM + ub0 + lc] = (__bf16)hfr0[i];
      h_bf[row * DIM + ub1 + lc] = (__bf16)hfr1[i];
    }
    if (threadIdx.x < 16) ybuf[threadIdx.x] = 0.0f;
    __syncthreads();

#pragma unroll
    for (int t = 0; t < 2; ++t) {
      const int ub  = t ? ub1 : ub0;
      const int col = ub + lc;                      // hidden-unit column 0..255
      v8f ghr = {}; v8f ghz = {}; v8f ghn = {};
#pragma unroll
      for (int k = 0; k < 8; ++k) {
        v16bf a  = load_a_rm(h_bf + k * 32, DIM);
        v16bf br = load_b_rm(Whhb + (0 * DIM + ub) * DIM + k * 32, DIM);
        v16bf bz = load_b_rm(Whhb + (1 * DIM + ub) * DIM + k * 32, DIM);
        v16bf bn = load_b_rm(Whhb + (2 * DIM + ub) * DIM + k * 32, DIM);
        ghr = __builtin_amdgcn_wmma_f32_16x16x32_bf16(false, a, false, br,
                                                      (short)0, ghr, false, false);
        ghz = __builtin_amdgcn_wmma_f32_16x16x32_bf16(false, a, false, bz,
                                                      (short)0, ghz, false, false);
        ghn = __builtin_amdgcn_wmma_f32_16x16x32_bf16(false, a, false, bn,
                                                      (short)0, ghn, false, false);
      }
      float bhr = b_hh[col];
      float bhz = b_hh[DIM + col];
      float bhn = b_hh[2 * DIM + col];
      float wo  = W_out[col];
#pragma unroll
      for (int i = 0; i < 8; ++i) {
        int  row  = i + (half << 3);
        long grow = ((long)(n0 + row) * SLATE_N + s) * THREE_D + col;
        float g1r = (float)G1[grow];
        float g1z = (float)G1[grow + DIM];
        float g1n = (float)G1[grow + 2 * DIM];
        float g2r = (float)G2[grow];
        float g2z = (float)G2[grow + DIM];
        float g2n = (float)G2[grow + 2 * DIM];
        float mi  = mbuf[row];
        float r   = sigmoidf_(g1r + mi * g2r + ghr[i] + bhr);
        float z   = sigmoidf_(g1z + mi * g2z + ghz[i] + bhz);
        float n   = tanhf(g1n + mi * g2n + r * (ghn[i] + bhn));
        float hold = t ? hfr1[i] : hfr0[i];
        float hnew = (1.0f - z) * n + z * hold;
        if (t) hfr1[i] = hnew; else hfr0[i] = hnew;
        atomicAdd(&ybuf[row], hnew * wo);
      }
    }
    __syncthreads();

    if (threadIdx.x < 16) {
      float y = ybuf[threadIdx.x] + bout;
      out[(long)(n0 + threadIdx.x) * SLATE_N + s] = y;   // pre-sigmoid, per ref
      mbuf[threadIdx.x] = sigmoidf_(y);                  // m for next step
    }
    __syncthreads();
  }
}

// ---- host launcher ---------------------------------------------------------
extern "C" void kernel_launch(void* const* d_in, const int* in_sizes, int n_in,
                              void* d_out, int out_size, void* d_ws, size_t ws_size,
                              hipStream_t stream) {
  const int*   item_idxs  = (const int*)  d_in[0];
  const int*   user_idxs  = (const int*)  d_in[1];
  // d_in[2] responses: unused by reference math
  const float* item_table = (const float*)d_in[3];
  const float* user_table = (const float*)d_in[4];
  const float* W_ih       = (const float*)d_in[5];
  const float* W_hh       = (const float*)d_in[6];
  const float* b_ih       = (const float*)d_in[7];
  const float* b_hh       = (const float*)d_in[8];
  const float* W_out      = (const float*)d_in[9];
  const float* b_out      = (const float*)d_in[10];
  float* out = (float*)d_out;

  char* p = (char*)d_ws;
  auto take = [&](size_t bytes) {
    char* q = p;
    p += (bytes + 255) & ~(size_t)255;
    return q;
  };
  __bf16* W1b  = (__bf16*)take((size_t)THREE_D * DIM * 2);
  __bf16* W2b  = (__bf16*)take((size_t)THREE_D * DIM * 2);
  __bf16* Whhb = (__bf16*)take((size_t)THREE_D * DIM * 2);
  __bf16* G1   = (__bf16*)take((size_t)NG * THREE_D * 2);   // ~126 MB
  __bf16* G2   = (__bf16*)take((size_t)NG * THREE_D * 2);   // ~126 MB

  k_prep_weights<<<(THREE_D * DIM + 255) / 256, 256, 0, stream>>>(
      W_ih, W_hh, W1b, W2b, Whhb);

  k_gemm_gi<<<NG / 16, 256, 0, stream>>>(
      item_idxs, item_table, b_ih, W1b, W2b, G1, G2);

  k_recurrence<<<NROWS / 16, 256, 0, stream>>>(
      user_idxs, user_table, b_hh, W_out, b_out, Whhb, G1, G2, out);
}
